// MambaBlock_66125316489621
// MI455X (gfx1250) — matrix operand order repaired and checked by reading, not compiled, not run on hardware
//
#include <hip/hip_runtime.h>
#include <hip/hip_bf16.h>
#include <math.h>

#define BATCH  2
#define SEQ    2048
#define DMODEL 1024
#define DINNER 2048
#define DCONV  4
#define DTRANK 64
#define DSTATE 16
#define NPROJ  96                  // DTRANK + 2*DSTATE
#define MTOK   (BATCH * SEQ)       // 4096 tokens

typedef __attribute__((ext_vector_type(16))) __bf16 v16bf;
typedef __attribute__((ext_vector_type(8)))  float  v8f;

union Frag { v16bf v; uint4 q[2]; };

__device__ __forceinline__ unsigned short f32_to_bf16_bits(float f) {
  unsigned int u = __float_as_uint(f);
  unsigned int r = (u + 0x7FFFu + ((u >> 16) & 1u)) >> 16;  // RNE
  return (unsigned short)r;
}

// ---------------------------------------------------------------- converters
__global__ void cvt_f32_bf16_kernel(const float* __restrict__ src,
                                    unsigned short* __restrict__ dst, int n) {
  int i = blockIdx.x * blockDim.x + threadIdx.x;
  if (i < n) dst[i] = f32_to_bf16_bits(src[i]);
}

// ---------------------------------------------------------------- WMMA GEMM
// C[M,N] = A[M,K] (row stride lda) * W[N,K]^T   (torch Linear weight layout)
// Block tile: 128(M) x 64(N). 8 waves; each wave owns a 16-row M slab and all
// four 16-col N tiles. The 64x32 B k-slice is staged in LDS (double-buffered,
// 40-element padded rows -> conflict-free 16B reads) and shared by all waves.
// k-loop is unrolled 2x with ping-pong A fragments / LDS buffers; a scheduler
// barrier pins the 8 ds_load_b128 ahead of the 4 back-to-back WMMAs.
// Requires: M % 128 == 0, K % 64 == 0 ... or K == 64 (steps even).
__global__ __launch_bounds__(256)
void wmma_gemm_bf16_lds(const __bf16* __restrict__ A, const __bf16* __restrict__ W,
                        float* __restrict__ C, int M, int N, int K, int lda) {
  constexpr int NT = 4;
  constexpr int BN = NT * 16;        // 64
  constexpr int STRIDE = 40;         // 32 k elements + 8 pad (80 B rows)
  __shared__ alignas(16) __bf16 lbs[2][BN * STRIDE];

  const int tid  = threadIdx.x;
  const int lane = tid & 31;
  const int wave = tid >> 5;
  const bool hi  = lane >= 16;
  const int l15  = lane & 15;

  const int nBlocks = (N + BN - 1) / BN;
  const int bm = blockIdx.x / nBlocks;
  const int bn = blockIdx.x % nBlocks;
  const int m0 = bm * 128 + wave * 16;

  // cooperative B loader: 256 threads = 64 rows x 4 chunks of 8 bf16 (16 B)
  const int lrow   = tid >> 2;             // 0..63
  const int lchunk = (tid & 3) * 8;        // k-offset 0,8,16,24
  int gn = bn * BN + lrow;
  const __bf16* gB = W + (size_t)(gn < N ? gn : 0) * K + lchunk;
  const int lsto = lrow * STRIDE + lchunk; // LDS store offset (elements)

  v8f acc[NT] = {};
  const __bf16* arow = A + (size_t)(m0 + l15) * lda;
  const int kaoff = hi ? 8 : 0;

  bool nok[NT];
#pragma unroll
  for (int t = 0; t < NT; ++t) nok[t] = (bn * BN + t * 16) < N;

  // one k-substep: B fragments from LDS (one clause), then 4 WMMAs
  auto mma_step = [&](const __bf16* lbuf, const Frag& a8) {
    Frag b[NT];
#pragma unroll
    for (int t = 0; t < NT; ++t) {
      const __bf16* lp = &lbuf[(t * 16 + l15) * STRIDE + (hi ? 8 : 0)];
      b[t].q[0] = *(const uint4*)(lp);
      b[t].q[1] = *(const uint4*)(lp + 16);
    }
    __builtin_amdgcn_sched_barrier(0);   // keep all ds_loads ahead of the WMMAs
#pragma unroll
    for (int t = 0; t < NT; ++t)
      acc[t] = __builtin_amdgcn_wmma_f32_16x16x32_bf16(
          false, a8.v, false, b[t].v, (short)0, acc[t], false, false);
  };

  // prime: B k-slice 0 into buffer 0, A fragment for k-step 0
  Frag aF[2];
  {
    uint4 v = *(const uint4*)(gB);
    *(uint4*)(&lbs[0][lsto]) = v;
    aF[0].q[0] = *(const uint4*)(arow + kaoff);
    aF[0].q[1] = *(const uint4*)(arow + kaoff + 16);
  }
  __syncthreads();

  const int steps = K / 32;                // even for all GEMMs used here
  for (int s = 0; s < steps; s += 2) {
    const int k0 = s * 32;

    // ---- substep s: compute from buf0/aF[0]; prefetch s+1 into buf1/aF[1]
    {
      uint4 nB = *(const uint4*)(gB + k0 + 32);
      int ka = k0 + 32 + kaoff;
      aF[1].q[0] = *(const uint4*)(arow + ka);
      aF[1].q[1] = *(const uint4*)(arow + ka + 16);
      mma_step(&lbs[0][0], aF[0]);
      *(uint4*)(&lbs[1][lsto]) = nB;
    }
    __syncthreads();

    // ---- substep s+1: compute from buf1/aF[1]; prefetch s+2 into buf0/aF[0]
    {
      const bool hn = (s + 2) < steps;
      uint4 nB;
      if (hn) {
        nB = *(const uint4*)(gB + k0 + 64);
        int ka = k0 + 64 + kaoff;
        aF[0].q[0] = *(const uint4*)(arow + ka);
        aF[0].q[1] = *(const uint4*)(arow + ka + 16);
      }
      mma_step(&lbs[1][0], aF[1]);
      if (hn) *(uint4*)(&lbs[0][lsto]) = nB;
    }
    __syncthreads();
  }

#pragma unroll
  for (int t = 0; t < NT; ++t) {
    if (!nok[t]) continue;
    int ncol = bn * BN + t * 16 + l15;
#pragma unroll
    for (int r = 0; r < 8; ++r) {
      int row = m0 + (hi ? 8 + r : r);       // C layout: lanes 16-31 hold M=8..15
      C[(size_t)row * N + ncol] = acc[t][r];
    }
  }
}

// ------------------------------------------------- depthwise causal conv+SiLU
__global__ void conv_silu_kernel(const float* __restrict__ xz,
                                 const float* __restrict__ cw,
                                 const float* __restrict__ cb,
                                 float* __restrict__ u,
                                 unsigned short* __restrict__ ubf) {
  int i = blockIdx.x * blockDim.x + threadIdx.x;
  if (i >= BATCH * SEQ * DINNER) return;
  int d = i % DINNER;
  int l = (i / DINNER) % SEQ;
  int b = i / (DINNER * SEQ);
  float acc = cb[d];
#pragma unroll
  for (int j = 0; j < DCONV; ++j) {
    int lj = l - (DCONV - 1) + j;
    if (lj >= 0)
      acc += xz[((size_t)(b * SEQ + lj)) * (2 * DINNER) + d] * cw[d * DCONV + j];
  }
  float s = acc / (1.0f + __expf(-acc));   // SiLU
  u[i]   = s;
  ubf[i] = f32_to_bf16_bits(s);
}

// ------------------------------------------------------------- dt = softplus
__global__ void dt_softplus_kernel(float* __restrict__ dt,
                                   const float* __restrict__ bdt, int n) {
  int i = blockIdx.x * blockDim.x + threadIdx.x;
  if (i >= n) return;
  float v = dt[i] + bdt[i % DINNER];
  dt[i] = (v > 20.0f) ? v : log1pf(__expf(v));
}

// ------------------------------------------------------------ selective scan
__global__ void scan_kernel(const float* __restrict__ dt,
                            const float* __restrict__ u,
                            const float* __restrict__ xdbl,
                            const float* __restrict__ Alog,
                            float* __restrict__ y) {
  int tid = blockIdx.x * blockDim.x + threadIdx.x;
  if (tid >= BATCH * DINNER) return;
  int b = tid / DINNER, d = tid % DINNER;
  float a[DSTATE], h[DSTATE];
#pragma unroll
  for (int n = 0; n < DSTATE; ++n) {
    a[n] = -__expf(Alog[d * DSTATE + n]);
    h[n] = 0.f;
  }
  const float* xd = xdbl + (size_t)b * SEQ * NPROJ;
  for (int l = 0; l < SEQ; ++l) {
    size_t idx = ((size_t)(b * SEQ + l)) * DINNER + d;
    float dtv = dt[idx];
    float du  = dtv * u[idx];
    const float* r = xd + (size_t)l * NPROJ;   // broadcast across wave lanes
    float acc = 0.f;
#pragma unroll
    for (int n = 0; n < DSTATE; ++n) {
      h[n] = __expf(dtv * a[n]) * h[n] + du * r[DTRANK + n];
      acc += h[n] * r[DTRANK + DSTATE + n];
    }
    y[idx] = acc;
  }
}

// --------------------------------------------------- gating: (y+u*D)*silu(z)
__global__ void gate_kernel(const float* __restrict__ ys,
                            const float* __restrict__ u,
                            const float* __restrict__ xz,
                            const float* __restrict__ Dp,
                            unsigned short* __restrict__ ybf) {
  int i = blockIdx.x * blockDim.x + threadIdx.x;
  if (i >= BATCH * SEQ * DINNER) return;
  int d = i % DINNER;
  size_t bl = (size_t)(i / DINNER);
  float z  = xz[bl * (2 * DINNER) + DINNER + d];
  float yv = ys[i] + u[i] * Dp[d];
  float s  = z / (1.0f + __expf(-z));
  ybf[i] = f32_to_bf16_bits(yv * s);
}

// ---------------------------------------------------------------------------
extern "C" void kernel_launch(void* const* d_in, const int* in_sizes, int n_in,
                              void* d_out, int out_size, void* d_ws, size_t ws_size,
                              hipStream_t stream) {
  (void)in_sizes; (void)n_in; (void)out_size; (void)ws_size;
  const float* x     = (const float*)d_in[0];
  const float* W_in  = (const float*)d_in[1];
  const float* cw    = (const float*)d_in[2];
  const float* cb    = (const float*)d_in[3];
  const float* W_xp  = (const float*)d_in[4];
  const float* W_dt  = (const float*)d_in[5];
  const float* b_dt  = (const float*)d_in[6];
  const float* A_log = (const float*)d_in[7];
  const float* Dp    = (const float*)d_in[8];
  const float* W_out = (const float*)d_in[9];
  float* out = (float*)d_out;

  char* ws = (char*)d_ws;
  size_t off = 0;
  auto carve = [&](size_t bytes) { size_t o = off; off += (bytes + 255) & ~(size_t)255; return o; };

  float* xz      = (float*)(ws + carve((size_t)MTOK * 2 * DINNER * 4)); // 64 MB
  float* u       = (float*)(ws + carve((size_t)MTOK * DINNER * 4));     // 32 MB
  float* xdbl    = (float*)(ws + carve((size_t)MTOK * NPROJ * 4));
  float* dt      = (float*)(ws + carve((size_t)MTOK * DINNER * 4));     // 32 MB
  float* yscan   = (float*)(ws + carve((size_t)MTOK * DINNER * 4));     // 32 MB
  unsigned short* xbf    = (unsigned short*)(ws + carve((size_t)MTOK * DMODEL * 2));
  unsigned short* winbf  = (unsigned short*)(ws + carve((size_t)(2 * DINNER) * DMODEL * 2));
  unsigned short* wxpbf  = (unsigned short*)(ws + carve((size_t)NPROJ * DINNER * 2));
  unsigned short* wdtbf  = (unsigned short*)(ws + carve((size_t)DINNER * DTRANK * 2));
  unsigned short* woutbf = (unsigned short*)(ws + carve((size_t)DMODEL * DINNER * 2));
  unsigned short* ubf    = (unsigned short*)(ws + carve((size_t)MTOK * DINNER * 2));
  unsigned short* xdblbf = (unsigned short*)(ws + carve((size_t)MTOK * NPROJ * 2));
  unsigned short* ybf    = (unsigned short*)(ws + carve((size_t)MTOK * DINNER * 2));

  auto cdiv = [](int a, int b) { return (a + b - 1) / b; };
  const int T = 256;

  // bf16 staging of activations and weights
  { int n = MTOK * DMODEL;          cvt_f32_bf16_kernel<<<cdiv(n, T), T, 0, stream>>>(x,     xbf,    n); }
  { int n = 2 * DINNER * DMODEL;    cvt_f32_bf16_kernel<<<cdiv(n, T), T, 0, stream>>>(W_in,  winbf,  n); }
  { int n = NPROJ * DINNER;         cvt_f32_bf16_kernel<<<cdiv(n, T), T, 0, stream>>>(W_xp,  wxpbf,  n); }
  { int n = DINNER * DTRANK;        cvt_f32_bf16_kernel<<<cdiv(n, T), T, 0, stream>>>(W_dt,  wdtbf,  n); }
  { int n = DMODEL * DINNER;        cvt_f32_bf16_kernel<<<cdiv(n, T), T, 0, stream>>>(W_out, woutbf, n); }

  // xz = x @ W_in^T   [4096 x 4096], K=1024
  {
    int blocks = (MTOK / 128) * cdiv(2 * DINNER, 64);   // 32 * 64 = 2048
    wmma_gemm_bf16_lds<<<blocks, T, 0, stream>>>(
        (const __bf16*)xbf, (const __bf16*)winbf, xz, MTOK, 2 * DINNER, DMODEL, DMODEL);
  }

  // depthwise causal conv + SiLU -> u (f32 + bf16)
  { int n = MTOK * DINNER; conv_silu_kernel<<<cdiv(n, T), T, 0, stream>>>(xz, cw, cb, u, ubf); }

  // x_dbl = u @ W_xproj^T  [4096 x 96], K=2048
  {
    int blocks = (MTOK / 128) * cdiv(NPROJ, 64);        // 32 * 2 = 64
    wmma_gemm_bf16_lds<<<blocks, T, 0, stream>>>(
        (const __bf16*)ubf, (const __bf16*)wxpbf, xdbl, MTOK, NPROJ, DINNER, DINNER);
  }

  // dt_raw = dt_lo @ W_dt^T  [4096 x 2048], K=64 (lda=96 into x_dbl)
  { int n = MTOK * NPROJ; cvt_f32_bf16_kernel<<<cdiv(n, T), T, 0, stream>>>(xdbl, xdblbf, n); }
  {
    int blocks = (MTOK / 128) * cdiv(DINNER, 64);       // 32 * 32 = 1024
    wmma_gemm_bf16_lds<<<blocks, T, 0, stream>>>(
        (const __bf16*)xdblbf, (const __bf16*)wdtbf, dt, MTOK, DINNER, DTRANK, NPROJ);
  }
  { int n = MTOK * DINNER; dt_softplus_kernel<<<cdiv(n, T), T, 0, stream>>>(dt, b_dt, n); }

  // selective scan (sequential in L, parallel over B*d_inner channels)
  { int n = BATCH * DINNER; scan_kernel<<<cdiv(n, T), T, 0, stream>>>(dt, u, xdbl, A_log, yscan); }

  // gating -> bf16
  { int n = MTOK * DINNER; gate_kernel<<<cdiv(n, T), T, 0, stream>>>(yscan, u, xz, Dp, ybf); }

  // out = y @ W_out^T  [4096 x 1024], K=2048
  {
    int blocks = (MTOK / 128) * cdiv(DMODEL, 64);       // 32 * 16 = 512
    wmma_gemm_bf16_lds<<<blocks, T, 0, stream>>>(
        (const __bf16*)ybf, (const __bf16*)woutbf, out, MTOK, DMODEL, DINNER, DINNER);
  }
}
